// GPT2Block_90666759618618
// MI455X (gfx1250) — matrix operand order repaired and checked by reading, not compiled, not run on hardware
//
#include <hip/hip_runtime.h>
#include <hip/hip_bf16.h>
#include <math.h>

// ---------------------------------------------------------------------------
// Types for CDNA5 WMMA / TDM (gfx1250, wave32)
// ---------------------------------------------------------------------------
typedef __attribute__((ext_vector_type(8)))  __bf16 v8bf;
typedef __attribute__((ext_vector_type(16))) __bf16 v16bf;
typedef __attribute__((ext_vector_type(8)))  float  v8f;
typedef __attribute__((ext_vector_type(4)))  unsigned int u32x4;
typedef __attribute__((ext_vector_type(8)))  int i32x8;
typedef __attribute__((ext_vector_type(4)))  int i32x4;

__device__ __forceinline__ v16bf cat16(v8bf lo, v8bf hi) {
  return __builtin_shufflevector(lo, hi, 0,1,2,3,4,5,6,7,8,9,10,11,12,13,14,15);
}

// A-matrix fragment (16x32 bf16): lane l holds row m=l%16; 16 contiguous K
// at (l/16)*16 (matches ldfragB k order).
__device__ __forceinline__ v16bf ldfragA(const __bf16* base, int half) {
  v8bf lo = *(const v8bf*)(base + half * 16);
  v8bf hi = *(const v8bf*)(base + half * 16 + 8);
  return cat16(lo, hi);
}

// B-matrix fragment (32x16 bf16) from K-contiguous (N-major) storage:
// lane l holds column n=l%16, 16 contiguous K at (l/16)*16 (pre-added).
__device__ __forceinline__ v16bf ldfragB(const __bf16* base) {
  v8bf lo = *(const v8bf*)(base);
  v8bf hi = *(const v8bf*)(base + 8);
  return cat16(lo, hi);
}

// Low 32 bits of a generic pointer to an LDS object == LDS byte offset
// (shared aperture places the offset in addr[31:0]).
__device__ __forceinline__ unsigned int lds_off32(const void* p) {
  return (unsigned int)(uintptr_t)p;
}

// ---------------------------------------------------------------------------
// Tensor Data Mover: 2-D tile load Global -> LDS (tensor_load_to_lds).
//   tile_d0 elements per row (contiguous, data_size=2B), tile_d1 rows,
//   row stride = stride_elems, LDS padding per D#.pad_* (padcfg).
// padcfg = (1<<20) pad_enable | (interval_code<<22) | (amount_code<<25)
// ---------------------------------------------------------------------------
__device__ __forceinline__ void tdm_load_2d(const __bf16* gptr,
                                            unsigned int lds_addr,
                                            unsigned int tensor_d0,
                                            unsigned int tensor_d1,
                                            unsigned int tile_d0,
                                            unsigned int tile_d1,
                                            unsigned long long stride_elems,
                                            unsigned int padcfg) {
  const unsigned long long ga = (unsigned long long)(uintptr_t)gptr;
  u32x4 g0;
  g0[0] = 1u;                                   // count=1, user-mode D#
  g0[1] = lds_addr;                             // lds_addr [63:32]
  g0[2] = (unsigned int)ga;                     // global_addr[31:0]
  g0[3] = (unsigned int)((ga >> 32) & 0x1FFFFFFull) | (2u << 30);  // [56:32], type=2
  i32x8 g1;
  g1[0] = (int)((1u << 16) | padcfg);           // data_size=2B + pad config
  g1[1] = (int)((tensor_d0 & 0xFFFFu) << 16);   // [47:32]=0, tensor_dim0 lo
  g1[2] = (int)((tensor_d0 >> 16) | ((tensor_d1 & 0xFFFFu) << 16));
  g1[3] = (int)((tensor_d1 >> 16) | (tile_d0 << 16));
  g1[4] = (int)tile_d1;                         // tile_dim1, tile_dim2=0
  g1[5] = (int)(unsigned int)(stride_elems & 0xFFFFFFFFull);
  g1[6] = (int)(unsigned int)((stride_elems >> 32) & 0xFFFFull);
  g1[7] = 0;
  const i32x4 z4 = {0, 0, 0, 0};
#if defined(__clang_major__) && (__clang_major__ >= 23)
  const i32x8 z8 = {0, 0, 0, 0, 0, 0, 0, 0};
  __builtin_amdgcn_tensor_load_to_lds(g0, g1, z4, z4, z8, 0);
#else
  __builtin_amdgcn_tensor_load_to_lds(g0, g1, z4, z4, 0);
#endif
}

// pad codes: interval k -> 2^(k+1) DWORDs before pad; amount k -> (k+1) DWORDs
#define PADCFG_ROW32  ((1u << 20) | (3u << 22) | (7u << 25))  // +32B per 64B row
#define PADCFG_ROW64  ((1u << 20) | (3u << 22) | (7u << 25))  // +32B per ... (row32 cfg used for 64B rows too)
#define PADCFG_ROW128 ((1u << 20) | (5u << 22) | (7u << 25))  // +32B per 256B row
#define PADCFG_KEY64  ((1u << 20) | (3u << 22) | (7u << 25))  // +32B per 128B? see V note

// ---------------------------------------------------------------------------
// Weight convert + transpose:  w[K][N] f32  ->  wt[N][K] bf16
// ---------------------------------------------------------------------------
__global__ void cvt_t_kernel(const float* __restrict__ w,
                             __bf16* __restrict__ wt, int K, int N) {
  const size_t idx = (size_t)blockIdx.x * 256 + threadIdx.x;  // over N*K
  const int n = (int)(idx / (size_t)K);
  const int k = (int)(idx % (size_t)K);
  wt[idx] = (__bf16)w[(size_t)k * N + n];
}

// ---------------------------------------------------------------------------
// V repack:  qkv[b*S+s][4096 + h*128 + d]  ->  vT[b][h][d][s]  (bf16)
// One-time 33MB so attention V tiles are TDM-loadable in B-frag layout.
// ---------------------------------------------------------------------------
__global__ void repack_v_kernel(const __bf16* __restrict__ qkv,
                                __bf16* __restrict__ vT) {
  const size_t idx = (size_t)blockIdx.x * 256 + threadIdx.x;  // [b][h][d][s]
  const int s = (int)(idx & 2047);
  const int d = (int)((idx >> 11) & 127);
  const int h = (int)((idx >> 18) & 15);
  const int b = (int)(idx >> 22);
  vT[idx] = qkv[(size_t)(b * 2048 + s) * 6144 + 4096 + h * 128 + d];
}

// ---------------------------------------------------------------------------
// LayerNorm (f32 in) -> bf16 out, one block per row of D=2048
// ---------------------------------------------------------------------------
__global__ __launch_bounds__(256) void ln_kernel(const float* __restrict__ x,
                                                 const float* __restrict__ g,
                                                 const float* __restrict__ bta,
                                                 __bf16* __restrict__ out) {
  __shared__ float red[16];
  const int row = blockIdx.x;
  const int tid = threadIdx.x, lane = tid & 31, wid = tid >> 5;
  const float* xr = x + (size_t)row * 2048;
  float v[8];
  float s = 0.f, s2 = 0.f;
#pragma unroll
  for (int k = 0; k < 8; ++k) {
    v[k] = xr[tid + k * 256];
    s += v[k];
    s2 += v[k] * v[k];
  }
#pragma unroll
  for (int off = 1; off < 32; off <<= 1) {
    s  += __shfl_xor(s,  off, 32);
    s2 += __shfl_xor(s2, off, 32);
  }
  if (lane == 0) { red[wid] = s; red[8 + wid] = s2; }
  __syncthreads();
  float ts = 0.f, ts2 = 0.f;
#pragma unroll
  for (int w = 0; w < 8; ++w) { ts += red[w]; ts2 += red[8 + w]; }
  const float mu  = ts  * (1.f / 2048.f);
  const float var = ts2 * (1.f / 2048.f) - mu * mu;
  const float inv = rsqrtf(var + 1e-5f);
  __bf16* orow = out + (size_t)row * 2048;
#pragma unroll
  for (int k = 0; k < 8; ++k) {
    const int col = tid + k * 256;
    orow[col] = (__bf16)((v[k] - mu) * inv * g[col] + bta[col]);
  }
}

// ---------------------------------------------------------------------------
// bf16 GEMM:  out = A[M][K] * Bt[N][K]^T  (+bias, optional gelu / residual)
// 128x128 block tile, BK=32, 256 threads = 8 waves, each wave 64x32.
// Global->LDS staging via Tensor Data Mover, double-buffered; wave 0 drives
// the TDM and TENSORcnt, block barrier publishes the buffer.
// MODE 0: bf16 out = acc + bias
// MODE 1: f32  out = acc + bias + residual
// MODE 2: bf16 out = gelu_tanh(acc + bias)
// ---------------------------------------------------------------------------
#define GBM 128
#define GBK 32
#define GLDK 48  // 32 + 16 pad (TDM pads 32B per 64B row) -> 96B stride

template <int MODE>
__global__ __launch_bounds__(256) void gemm_bf16_kernel(
    const __bf16* __restrict__ A, const __bf16* __restrict__ Bt,
    const float* __restrict__ bias, const float* __restrict__ res,
    float* __restrict__ out32, __bf16* __restrict__ outbf, int K, int N) {
  __shared__ __bf16 sA[2][GBM * GLDK];
  __shared__ __bf16 sB[2][GBM * GLDK];

  const int tid  = threadIdx.x;
  const int lane = tid & 31, wid = tid >> 5;
  const int wm = wid & 1, wn = wid >> 1;       // 2 x 4 wave grid
  const int half = lane >> 4, lr = lane & 15;
  const int m0 = blockIdx.y * GBM, n0 = blockIdx.x * GBM;

  const __bf16* Abase = A  + (size_t)m0 * K;
  const __bf16* Bbase = Bt + (size_t)n0 * K;
  const int nT = K / GBK;

  if (wid == 0) {  // prologue: stage k-tile 0 into buffer 0
    tdm_load_2d(Abase, lds_off32(&sA[0][0]), (unsigned)K, 1u << 20, GBK, GBM,
                (unsigned long long)K, PADCFG_ROW32);
    tdm_load_2d(Bbase, lds_off32(&sB[0][0]), (unsigned)K, 1u << 20, GBK, GBM,
                (unsigned long long)K, PADCFG_ROW32);
  }

  v8f acc[4][2];
#pragma unroll
  for (int fm = 0; fm < 4; ++fm)
#pragma unroll
    for (int fn = 0; fn < 2; ++fn) acc[fm][fn] = (v8f)(0.f);

  for (int t = 0; t < nT; ++t) {
    const int buf = t & 1;
    if (wid == 0) {
      if (t + 1 < nT) {  // async-stage next k-tile into other buffer
        const int kn = (t + 1) * GBK;
        tdm_load_2d(Abase + kn, lds_off32(&sA[buf ^ 1][0]), (unsigned)K,
                    1u << 20, GBK, GBM, (unsigned long long)K, PADCFG_ROW32);
        tdm_load_2d(Bbase + kn, lds_off32(&sB[buf ^ 1][0]), (unsigned)K,
                    1u << 20, GBK, GBM, (unsigned long long)K, PADCFG_ROW32);
        // TDM ops complete in-order: <=2 outstanding => current buffer landed
        __builtin_amdgcn_s_wait_tensorcnt(2);
      } else {
        __builtin_amdgcn_s_wait_tensorcnt(0);
      }
    }
    __syncthreads();  // publish staged buffer to all 8 waves

    v16bf af[4], bfr[2];
#pragma unroll
    for (int fm = 0; fm < 4; ++fm)
      af[fm] = ldfragA(&sA[buf][(wm * 64 + fm * 16 + lr) * GLDK], half);
#pragma unroll
    for (int fn = 0; fn < 2; ++fn)
      bfr[fn] = ldfragB(&sB[buf][(wn * 32 + fn * 16 + lr) * GLDK + half * 16]);

#pragma unroll
    for (int fm = 0; fm < 4; ++fm)
#pragma unroll
      for (int fn = 0; fn < 2; ++fn)
        acc[fm][fn] = __builtin_amdgcn_wmma_f32_16x16x32_bf16(
            false, af[fm], false, bfr[fn], (short)0, acc[fm][fn], false, false);
    __syncthreads();  // all reads done before this buffer is re-staged
  }

  // epilogue (C layout: VGPR i, lane l -> m = i + 8*(l/16), n = l%16)
#pragma unroll
  for (int fn = 0; fn < 2; ++fn) {
    const int col = n0 + wn * 32 + fn * 16 + lr;
    const float bv = bias[col];
#pragma unroll
    for (int fm = 0; fm < 4; ++fm) {
#pragma unroll
      for (int i = 0; i < 8; ++i) {
        const int row = m0 + wm * 64 + fm * 16 + i + 8 * half;
        float v = acc[fm][fn][i] + bv;
        if (MODE == 2) {
          const float u = v;
          v = 0.5f * u *
              (1.0f + tanhf(0.7978845608028654f * (u + 0.044715f * u * u * u)));
        }
        const size_t off = (size_t)row * N + col;
        if (MODE == 1) {
          out32[off] = v + res[off];
        } else {
          outbf[off] = (__bf16)v;
        }
      }
    }
  }
}

// ---------------------------------------------------------------------------
// Flash attention (causal), bf16 WMMA, fp32 softmax state.
// Grid: (S/128, H, B). 256 threads = 8 waves; wave w owns 16 q rows.
// K tile (64 keys x 128) staged by TDM in natural [key][d] layout (score
// B-frag native); V tile staged by TDM from the pre-transposed vT buffer as
// [d=128 rows][key=64], which is the PV B-frag-native layout. All LDS reads
// are vector ds_load_b128, fully compiler-scheduled. P goes C->A via LDS.
// qkv layout: bf16 [B*S][6144], q at +0, k at +2048 per row.
// ---------------------------------------------------------------------------
#define SKD 144  // 128 + 16 pad (TDM pads 32B per 256B row)
#define SVD 80   // 64 + 16 pad (TDM pads 32B per 128B row)
#define SPD 72   // 64 + 8 pad

__global__ __launch_bounds__(256) void attn_kernel(
    const __bf16* __restrict__ qkv, const __bf16* __restrict__ vT,
    __bf16* __restrict__ attnOut) {
  __shared__ __bf16 sK[64 * SKD];    // K tile, natural [key][d]
  __shared__ __bf16 sV[128 * SVD];   // V tile, transposed [d][key]
  __shared__ __bf16 sP[8 * 16 * SPD];

  const int tid = threadIdx.x, lane = tid & 31, w = tid >> 5;
  const int half = lane >> 4, lr = lane & 15;
  const int h = blockIdx.y, b = blockIdx.z;
  const int q0 = blockIdx.x * 128;

  // resident Q fragments for this wave (16 rows x 128 = 4 A-frags)
  const int qr = q0 + w * 16 + lr;
  const __bf16* qp = qkv + (size_t)(b * 2048 + qr) * 6144 + h * 128;
  v16bf qf[4];
#pragma unroll
  for (int kk = 0; kk < 4; ++kk) qf[kk] = ldfragA(qp + kk * 32, half);

  v8f o[8];
#pragma unroll
  for (int nd = 0; nd < 8; ++nd) o[nd] = (v8f)(0.f);
  float mx[8], li[8];
#pragma unroll
  for (int i = 0; i < 8; ++i) { mx[i] = -1e30f; li[i] = 0.f; }
  const float scale = 0.08838834764831845f;  // 1/sqrt(128)

  for (int j0 = 0; j0 < q0 + 128; j0 += 64) {
    __syncthreads();  // previous tile fully consumed
    if (w == 0) {     // stage K and V tiles via TDM
      const __bf16* kb = qkv + (size_t)(b * 2048 + j0) * 6144 + 2048 + h * 128;
      tdm_load_2d(kb, lds_off32(&sK[0]), 6144u, 8192u, 128, 64, 6144ull,
                  PADCFG_ROW128);
      const __bf16* vb = vT + (size_t)(b * 16 + h) * 128 * 2048 + j0;
      tdm_load_2d(vb, lds_off32(&sV[0]), 2048u, 8192u, 64, 128, 2048ull,
                  PADCFG_ROW32);
      __builtin_amdgcn_s_wait_tensorcnt(0);
    }
    __syncthreads();  // publish K/V tiles

    // scores S = Q K^T  (16 x 64 per wave)
    v8f s[4];
#pragma unroll
    for (int nf = 0; nf < 4; ++nf) s[nf] = (v8f)(0.f);
#pragma unroll
    for (int nf = 0; nf < 4; ++nf)
#pragma unroll
      for (int kk = 0; kk < 4; ++kk)
        s[nf] = __builtin_amdgcn_wmma_f32_16x16x32_bf16(
            false, qf[kk], false,
            ldfragB(&sK[(nf * 16 + lr) * SKD + kk * 32 + half * 16]), (short)0,
            s[nf], false, false);

    // causal mask + online softmax; rows live across 16-lane halves
    __bf16* sPw = &sP[(w * 16) * SPD];
#pragma unroll
    for (int i = 0; i < 8; ++i) {
      const int srow = q0 + w * 16 + i + 8 * half;
      float mloc = -1e30f;
#pragma unroll
      for (int nf = 0; nf < 4; ++nf) {
        const int scol = j0 + nf * 16 + lr;
        float sc = s[nf][i] * scale;
        sc = (scol <= srow) ? sc : -1e30f;
        s[nf][i] = sc;
        mloc = fmaxf(mloc, sc);
      }
#pragma unroll
      for (int off = 1; off < 16; off <<= 1)
        mloc = fmaxf(mloc, __shfl_xor(mloc, off, 16));
      const float mn = fmaxf(mx[i], mloc);
      const float al = __expf(mx[i] - mn);
      mx[i] = mn;
      float rs = 0.f;
#pragma unroll
      for (int nf = 0; nf < 4; ++nf) {
        const float p = __expf(s[nf][i] - mn);
        rs += p;
        sPw[(i + 8 * half) * SPD + nf * 16 + lr] = (__bf16)p;  // C->A via LDS
      }
#pragma unroll
      for (int off = 1; off < 16; off <<= 1) rs += __shfl_xor(rs, off, 16);
      li[i] = li[i] * al + rs;
#pragma unroll
      for (int nd = 0; nd < 8; ++nd) o[nd][i] *= al;
    }

    // O += P V  (per-wave LDS region; same-wave DS ops are in-order)
#pragma unroll
    for (int ks = 0; ks < 2; ++ks) {
      const v16bf pf = ldfragA(&sPw[lr * SPD + ks * 32], half);
#pragma unroll
      for (int nd = 0; nd < 8; ++nd)
        o[nd] = __builtin_amdgcn_wmma_f32_16x16x32_bf16(
            false, pf, false,
            ldfragB(&sV[(nd * 16 + lr) * SVD + ks * 32 + half * 16]), (short)0,
            o[nd], false, false);
    }
  }

  // normalize and write bf16 attn output [B*S][2048]
#pragma unroll
  for (int i = 0; i < 8; ++i) {
    const float inv = 1.f / li[i];
    const int srow = q0 + w * 16 + i + 8 * half;
    __bf16* orow = attnOut + (size_t)(b * 2048 + srow) * 2048 + h * 128;
#pragma unroll
    for (int nd = 0; nd < 8; ++nd)
      orow[nd * 16 + lr] = (__bf16)(o[nd][i] * inv);
  }
}

// ---------------------------------------------------------------------------
// Host-side orchestration
// ---------------------------------------------------------------------------
extern "C" void kernel_launch(void* const* d_in, const int* in_sizes, int n_in,
                              void* d_out, int out_size, void* d_ws,
                              size_t ws_size, hipStream_t stream) {
  (void)in_sizes; (void)n_in; (void)out_size; (void)ws_size;
  const float* hs        = (const float*)d_in[0];   // [4,2048,2048]
  const float* ln1_g     = (const float*)d_in[1];
  const float* ln1_b     = (const float*)d_in[2];
  const float* w_attn    = (const float*)d_in[3];   // [2048,6144]
  const float* b_attn    = (const float*)d_in[4];
  const float* w_o       = (const float*)d_in[5];   // [2048,2048]
  const float* b_o       = (const float*)d_in[6];
  const float* ln2_g     = (const float*)d_in[7];
  const float* ln2_b     = (const float*)d_in[8];
  const float* w_fc      = (const float*)d_in[9];   // [2048,8192]
  const float* b_fc      = (const float*)d_in[10];
  const float* w_fc_proj = (const float*)d_in[11];  // [8192,2048]
  const float* b_fc_proj = (const float*)d_in[12];

  const int M = 8192;  // B*S
  char* ws = (char*)d_ws;
  size_t off = 0;
  __bf16* wt_attn = (__bf16*)(ws + off); off += (size_t)6144 * 2048 * 2;
  __bf16* wt_o    = (__bf16*)(ws + off); off += (size_t)2048 * 2048 * 2;
  __bf16* wt_fc   = (__bf16*)(ws + off); off += (size_t)8192 * 2048 * 2;
  __bf16* wt_proj = (__bf16*)(ws + off); off += (size_t)2048 * 8192 * 2;
  __bf16* xln     = (__bf16*)(ws + off); off += (size_t)M * 2048 * 2;
  float*  hidden  = (float*) (ws + off); off += (size_t)M * 2048 * 4;
  __bf16* qkv     = (__bf16*)(ws + off); off += (size_t)M * 6144 * 2;
  __bf16* attnb   = (__bf16*)(ws + off); off += (size_t)M * 2048 * 2;
  __bf16* vT      = (__bf16*)(ws + off); off += (size_t)M * 2048 * 2;
  // hfc [M][8192] bf16 aliases qkv+attn region (both dead by the time FC1
  // runs); vT sits after attnb so it is untouched by hfc.
  __bf16* hfc = qkv;

  // 1) weights -> bf16, transposed [N][K]
  cvt_t_kernel<<<(6144 * 2048) / 256, 256, 0, stream>>>(w_attn, wt_attn, 2048, 6144);
  cvt_t_kernel<<<(2048 * 2048) / 256, 256, 0, stream>>>(w_o, wt_o, 2048, 2048);
  cvt_t_kernel<<<(8192 * 2048) / 256, 256, 0, stream>>>(w_fc, wt_fc, 2048, 8192);
  cvt_t_kernel<<<(2048 * 8192) / 256, 256, 0, stream>>>(w_fc_proj, wt_proj, 8192, 2048);

  // 2) LN1
  ln_kernel<<<M, 256, 0, stream>>>(hs, ln1_g, ln1_b, xln);

  // 3) QKV = xln @ w_attn + b_attn  -> bf16 [M][6144]
  gemm_bf16_kernel<0><<<dim3(6144 / 128, M / 128), 256, 0, stream>>>(
      xln, wt_attn, b_attn, nullptr, nullptr, qkv, 2048, 6144);

  // 4) repack V into [b][h][d][s] so attention V tiles are TDM-native
  repack_v_kernel<<<(4 * 16 * 128 * 2048) / 256, 256, 0, stream>>>(qkv, vT);

  // 5) causal flash attention -> bf16 [M][2048]
  attn_kernel<<<dim3(2048 / 128, 16, 4), 256, 0, stream>>>(qkv, vT, attnb);

  // 6) hidden = attn @ w_o + b_o + hs  -> f32 [M][2048]
  gemm_bf16_kernel<1><<<dim3(2048 / 128, M / 128), 256, 0, stream>>>(
      attnb, wt_o, b_o, hs, hidden, nullptr, 2048, 2048);

  // 7) LN2
  ln_kernel<<<M, 256, 0, stream>>>(hidden, ln2_g, ln2_b, xln);

  // 8) hfc = gelu(xln @ w_fc + b_fc) -> bf16 [M][8192]
  gemm_bf16_kernel<2><<<dim3(8192 / 128, M / 128), 256, 0, stream>>>(
      xln, wt_fc, b_fc, nullptr, nullptr, hfc, 2048, 8192);

  // 9) out = hfc @ w_fc_proj + b_fc_proj + hidden -> f32 [M][2048]
  gemm_bf16_kernel<1><<<dim3(2048 / 128, M / 128), 256, 0, stream>>>(
      hfc, wt_proj, b_fc_proj, hidden, (float*)d_out, nullptr, 8192, 2048);
}
